// Mamba2Block_16544214024872
// MI455X (gfx1250) — compile-verified, matrix operand order checked by reading, not run on hardware
//
#include <hip/hip_runtime.h>

// ---------------- problem constants ----------------
#define BSZ      2
#define SEQ      2048
#define DMODEL   1024
#define DINNER   2048
#define DSTATE   16
#define DCONV    4
#define NROWS    (BSZ * SEQ)              // 4096
#define XDBL_N   (DINNER + 2 * DSTATE)    // 2080

// ---------------- vector types ----------------
typedef __attribute__((ext_vector_type(16))) __bf16 bf16x16;
typedef __attribute__((ext_vector_type(8)))  __bf16 bf16x8;
typedef __attribute__((ext_vector_type(8)))  float  f32x8;
typedef __attribute__((ext_vector_type(4)))  float  f32x4;
typedef int i32x4v __attribute__((vector_size(16)));   // matches builtin's pointee type

// ---------------- gfx1250 async-to-LDS path (guarded) ----------------
#if defined(__HIP_DEVICE_COMPILE__) && __has_builtin(__builtin_amdgcn_global_load_async_to_lds_b128)
#define USE_ASYNC_LDS 1
#else
#define USE_ASYNC_LDS 0
#endif

#if USE_ASYNC_LDS
__device__ __forceinline__ void async_copy_b128(const void* g, void* l) {
  auto gp = (__attribute__((address_space(1))) i32x4v*)(unsigned long long)g;
  auto lp = (__attribute__((address_space(3))) i32x4v*)(unsigned)(unsigned long long)l;
  __builtin_amdgcn_global_load_async_to_lds_b128(gp, lp, 0, 0);
}
__device__ __forceinline__ void wait_async_lds() {
#if __has_builtin(__builtin_amdgcn_s_wait_asynccnt)
  __builtin_amdgcn_s_wait_asynccnt(0);
#else
  asm volatile("s_wait_asynccnt 0" ::: "memory");
#endif
}
#endif

// ---------------- math helpers ----------------
__device__ __forceinline__ float sigmoid_f(float x) { return 1.0f / (1.0f + __expf(-x)); }
__device__ __forceinline__ float silu_f(float x)    { return x * sigmoid_f(x); }
__device__ __forceinline__ float softplus_f(float x){ return fmaxf(x, 0.0f) + log1pf(__expf(-fabsf(x))); }

// ---------------- weight convert + transpose: W[K][N] f32 -> WT[N][K] bf16 ----------------
__global__ __launch_bounds__(256) void wcvt_t(const float* __restrict__ W,
                                              __bf16* __restrict__ WT, int K, int N) {
  int idx = blockIdx.x * 256 + threadIdx.x;
  if (idx < K * N) {
    int k = idx / N;
    int n = idx - k * N;
    WT[(size_t)n * K + k] = (__bf16)W[idx];
  }
}

// ---------------- bf16 WMMA GEMM: C[M][N] = A[M][K](f32, lda) * Bt[N][K](bf16) ----------------
// EPI==1: C = softplus(C + bias[n])
template <int EPI>
__global__ __launch_bounds__(256) void gemm_bf16_wmma(
    const float* __restrict__ A, int lda,
    const __bf16* __restrict__ Bt,
    float* __restrict__ C, int ldc,
    const float* __restrict__ bias,
    int M, int N, int K) {
  __shared__ __bf16 As[128][32];
  __shared__ __bf16 Bs[128][32];   // N-major: Bs[n][k]

  const int tid  = threadIdx.x;
  const int lane = tid & 31;
  const int wave = tid >> 5;       // 0..7
  const int wm   = wave >> 1;      // 0..3  -> 32-row band
  const int wn   = wave & 1;       // 0..1  -> 64-col band
  const int m0   = blockIdx.y * 128;
  const int n0   = blockIdx.x * 128;

  const int ldrow = tid >> 1;         // 0..127
  const int ldseg = (tid & 1) * 16;   // 0 or 16

  f32x8 acc[2][4] = {};

  for (int k0 = 0; k0 < K; k0 += 32) {
    // ---- stage A tile: f32 global -> bf16 LDS ----
    const float* ap = A + (size_t)(m0 + ldrow) * lda + k0 + ldseg;
    __builtin_prefetch(ap + 32, 0, 1);   // next K tile -> global_prefetch_b8
#pragma unroll
    for (int j = 0; j < 16; j += 4) {
      f32x4 v = *(const f32x4*)(ap + j);
      As[ldrow][ldseg + j + 0] = (__bf16)v[0];
      As[ldrow][ldseg + j + 1] = (__bf16)v[1];
      As[ldrow][ldseg + j + 2] = (__bf16)v[2];
      As[ldrow][ldseg + j + 3] = (__bf16)v[3];
    }
    // ---- stage B tile: bf16 global (N-major) -> LDS ----
    const int ncol = n0 + ldrow;
    if (ncol < N) {
      const __bf16* bp = Bt + (size_t)ncol * K + k0 + ldseg;
#if USE_ASYNC_LDS
      async_copy_b128(bp,     &Bs[ldrow][ldseg]);
      async_copy_b128(bp + 8, &Bs[ldrow][ldseg + 8]);
#else
      *(bf16x8*)&Bs[ldrow][ldseg]     = *(const bf16x8*)bp;
      *(bf16x8*)&Bs[ldrow][ldseg + 8] = *(const bf16x8*)(bp + 8);
#endif
    } else {
      bf16x8 zv = (bf16x8)((__bf16)0.0f);
      *(bf16x8*)&Bs[ldrow][ldseg]     = zv;
      *(bf16x8*)&Bs[ldrow][ldseg + 8] = zv;
    }
#if USE_ASYNC_LDS
    wait_async_lds();
#endif
    __syncthreads();

    // ---- WMMA compute: 2x4 tiles of 16x16 per wave ----
    {
      const int hi = lane >> 4;   // K-half select per 16-bit A/B layout
      const int ml = lane & 15;   // row (A) / col (B) within tile
      bf16x16 afr[2];
#pragma unroll
      for (int tm = 0; tm < 2; ++tm) {
        const __bf16* a0 = &As[wm * 32 + tm * 16 + ml][hi * 8];
        bf16x8 lo = *(const bf16x8*)a0;
        bf16x8 hp = *(const bf16x8*)(a0 + 16);
        afr[tm] = __builtin_shufflevector(lo, hp, 0, 1, 2, 3, 4, 5, 6, 7,
                                          8, 9, 10, 11, 12, 13, 14, 15);
      }
#pragma unroll
      for (int tn = 0; tn < 4; ++tn) {
        const __bf16* b0 = &Bs[wn * 64 + tn * 16 + ml][hi * 8];
        bf16x8 lo = *(const bf16x8*)b0;
        bf16x8 hp = *(const bf16x8*)(b0 + 16);
        bf16x16 bfr = __builtin_shufflevector(lo, hp, 0, 1, 2, 3, 4, 5, 6, 7,
                                              8, 9, 10, 11, 12, 13, 14, 15);
#pragma unroll
        for (int tm = 0; tm < 2; ++tm) {
          acc[tm][tn] = __builtin_amdgcn_wmma_f32_16x16x32_bf16(
              false, afr[tm], false, bfr, (short)0, acc[tm][tn], false, false);
        }
      }
    }
    __syncthreads();
  }

  // ---- store C (f32 C/D layout: M = r + 8*(lane>=16), N = lane&15) ----
  const int hi = lane >> 4;
  const int ml = lane & 15;
#pragma unroll
  for (int tm = 0; tm < 2; ++tm) {
#pragma unroll
    for (int tn = 0; tn < 4; ++tn) {
      const int col = n0 + wn * 64 + tn * 16 + ml;
      if (col < N) {
        const float bv = (EPI == 1) ? bias[col] : 0.0f;
#pragma unroll
        for (int r = 0; r < 8; ++r) {
          const int row = m0 + wm * 32 + tm * 16 + hi * 8 + r;
          float v = acc[tm][tn][r];
          if (EPI == 1) v = softplus_f(v + bv);
          C[(size_t)row * ldc + col] = v;
        }
      }
    }
  }
}

// ---------------- causal depthwise conv1d + SiLU ----------------
// xs = xz[:, :, 0:DINNER] (row stride 2*DINNER); u[b,l,d] = silu(conv(xs)+b)
__global__ __launch_bounds__(256) void conv_silu_kernel(
    const float* __restrict__ xz, const float* __restrict__ conv_w,
    const float* __restrict__ conv_b, float* __restrict__ u) {
  const int idx = blockIdx.x * 256 + threadIdx.x;   // over B*L*DINNER
  const int d = idx & (DINNER - 1);
  const int l = (idx >> 11) & (SEQ - 1);
  const int b = idx >> 22;
  const float* cw = conv_w + d * DCONV;
  float acc = conv_b[d];
#pragma unroll
  for (int j = 0; j < DCONV; ++j) {
    const int li = l - (DCONV - 1) + j;
    if (li >= 0)
      acc += xz[((size_t)(b * SEQ + li)) * (2 * DINNER) + d] * cw[j];
  }
  u[idx] = silu_f(acc);
}

// ---------------- selective scan: one lane per (b, d) channel ----------------
__global__ __launch_bounds__(256) void scan_kernel(
    const float* __restrict__ xz, const float* __restrict__ u,
    const float* __restrict__ x_dbl, const float* __restrict__ delta,
    const float* __restrict__ A_log, const float* __restrict__ D_param,
    float* __restrict__ y) {
  const int gid = blockIdx.x * 256 + threadIdx.x;   // 0..4095
  const int b = gid >> 11;
  const int d = gid & (DINNER - 1);
  float Aa[DSTATE], h[DSTATE];
#pragma unroll
  for (int n = 0; n < DSTATE; ++n) {
    Aa[n] = -__expf(A_log[d * DSTATE + n]);
    h[n] = 0.0f;
  }
  const float Dp = D_param[d];
  for (int l = 0; l < SEQ; ++l) {
    const size_t row = (size_t)(b * SEQ + l);
    const float dt = delta[row * DINNER + d];
    const float uu = u[row * DINNER + d];
    const float* bc = x_dbl + row * XDBL_N + DINNER;  // B then C, 16 each
    float yv = 0.0f;
#pragma unroll
    for (int n = 0; n < DSTATE; ++n) {
      const float dA = __expf(dt * Aa[n]);
      h[n] = dA * h[n] + dt * bc[n] * uu;
      yv += h[n] * bc[DSTATE + n];
    }
    const float zz = xz[row * (2 * DINNER) + DINNER + d];
    y[row * DINNER + d] = (yv + uu * Dp) * silu_f(zz);
  }
}

// ---------------- host launch ----------------
extern "C" void kernel_launch(void* const* d_in, const int* in_sizes, int n_in,
                              void* d_out, int out_size, void* d_ws, size_t ws_size,
                              hipStream_t stream) {
  const float* x       = (const float*)d_in[0];
  const float* w_in    = (const float*)d_in[1];
  const float* conv_w  = (const float*)d_in[2];
  const float* conv_b  = (const float*)d_in[3];
  const float* w_x     = (const float*)d_in[4];
  const float* w_dt    = (const float*)d_in[5];
  const float* b_dt    = (const float*)d_in[6];
  const float* A_log   = (const float*)d_in[7];
  const float* D_param = (const float*)d_in[8];
  const float* w_out   = (const float*)d_in[9];

  char* ws = (char*)d_ws;
  float* xz    = (float*)ws; ws += (size_t)NROWS * (2 * DINNER) * sizeof(float);
  float* u     = (float*)ws; ws += (size_t)NROWS * DINNER * sizeof(float);
  float* xdbl  = (float*)ws; ws += (size_t)NROWS * XDBL_N * sizeof(float);
  float* delta = (float*)ws; ws += (size_t)NROWS * DINNER * sizeof(float);
  float* y     = (float*)ws; ws += (size_t)NROWS * DINNER * sizeof(float);
  __bf16* w_inT  = (__bf16*)ws; ws += (size_t)(2 * DINNER) * DMODEL * sizeof(__bf16);
  __bf16* w_xT   = (__bf16*)ws; ws += (size_t)XDBL_N * DINNER * sizeof(__bf16);
  __bf16* w_dtT  = (__bf16*)ws; ws += (size_t)DINNER * DINNER * sizeof(__bf16);
  __bf16* w_outT = (__bf16*)ws; ws += (size_t)DMODEL * DINNER * sizeof(__bf16);

  // 1) weights -> bf16, transposed to [N][K]
  {
    int e;
    e = DMODEL * 2 * DINNER; wcvt_t<<<(e + 255) / 256, 256, 0, stream>>>(w_in,  w_inT,  DMODEL, 2 * DINNER);
    e = DINNER * XDBL_N;     wcvt_t<<<(e + 255) / 256, 256, 0, stream>>>(w_x,   w_xT,   DINNER, XDBL_N);
    e = DINNER * DINNER;     wcvt_t<<<(e + 255) / 256, 256, 0, stream>>>(w_dt,  w_dtT,  DINNER, DINNER);
    e = DINNER * DMODEL;     wcvt_t<<<(e + 255) / 256, 256, 0, stream>>>(w_out, w_outT, DINNER, DMODEL);
  }

  // 2) xz = x @ w_in                              (4096 x 4096 x 1024)
  gemm_bf16_wmma<0><<<dim3((2 * DINNER) / 128, NROWS / 128), 256, 0, stream>>>(
      x, DMODEL, w_inT, xz, 2 * DINNER, nullptr, NROWS, 2 * DINNER, DMODEL);

  // 3) u = silu(depthwise causal conv(xs) + conv_b)
  conv_silu_kernel<<<(NROWS * DINNER) / 256, 256, 0, stream>>>(xz, conv_w, conv_b, u);

  // 4) x_dbl = u @ w_x                            (4096 x 2080 x 2048)
  gemm_bf16_wmma<0><<<dim3((XDBL_N + 127) / 128, NROWS / 128), 256, 0, stream>>>(
      u, DINNER, w_xT, xdbl, XDBL_N, nullptr, NROWS, XDBL_N, DINNER);

  // 5) delta = softplus(x_dbl[:, :DINNER] @ w_dt + b_dt)
  gemm_bf16_wmma<1><<<dim3(DINNER / 128, NROWS / 128), 256, 0, stream>>>(
      xdbl, XDBL_N, w_dtT, delta, DINNER, b_dt, NROWS, DINNER, DINNER);

  // 6) selective scan + skip (u*D) + gate silu(z)
  scan_kernel<<<(BSZ * DINNER) / 256, 256, 0, stream>>>(xz, u, xdbl, delta, A_log, D_param, y);

  // 7) out = y @ w_out                            (4096 x 1024 x 2048)
  gemm_bf16_wmma<0><<<dim3(DMODEL / 128, NROWS / 128), 256, 0, stream>>>(
      y, DINNER, w_outT, (float*)d_out, DMODEL, nullptr, NROWS, DMODEL, DINNER);
}
